// FixedPointTransformerEncoderLayer_56942676410840
// MI455X (gfx1250) — matrix operand order repaired and checked by reading, not compiled, not run on hardware
//
#include <hip/hip_runtime.h>

typedef _Float16 f16;
typedef __attribute__((ext_vector_type(16))) _Float16 v16h;
typedef __attribute__((ext_vector_type(8)))  _Float16 v8h;
typedef __attribute__((ext_vector_type(8)))  float    v8f;
typedef __attribute__((ext_vector_type(4)))  unsigned u32x4;
typedef __attribute__((ext_vector_type(8)))  int      i32x8;
typedef __attribute__((ext_vector_type(4)))  int      i32x4;

__device__ __forceinline__ float fpq1(float v) {
    float r = rintf(v * 256.0f);
    r = fminf(fmaxf(r, -32768.0f), 32767.0f);
    return r * 0.00390625f;
}

__device__ __forceinline__ v8f wmma16(v16h a, v16h b, v8f c) {
    return __builtin_amdgcn_wmma_f32_16x16x32_f16(false, a, false, b, (short)0, c,
                                                  false, false);
}

__device__ __forceinline__ v16h pack2(v8h lo, v8h hi) {
    v16h o;
#pragma unroll
    for (int j = 0; j < 8; ++j) { o[j] = lo[j]; o[8 + j] = hi[j]; }
    return o;
}

// ---- CDNA5 async memory->LDS copy (ASYNCcnt-tracked), 16 bytes per lane ----
// LDS byte address = low 32 bits of the generic pointer (aperture rule, ISA 10.2).
__device__ __forceinline__ void async_ld16(void* lds_dst, const void* gsrc) {
    unsigned off = (unsigned)(unsigned long long)lds_dst;
    asm volatile("global_load_async_to_lds_b128 %0, %1, off"
                 :: "v"(off), "v"(gsrc) : "memory");
}
__device__ __forceinline__ void wait_async0() {
    asm volatile("s_wait_asynccnt 0" ::: "memory");
}

// ---- CDNA5 Tensor Data Mover: 2D f16 tile DMA global -> LDS --------------
// D# per ISA 8.3/8.4: group0 = {flags, lds_addr, global_addr, type=2},
// group1 carries data_size/pad config, tensor dims (set large: tiles are
// always in-bounds here), tile dims, and the row stride in elements.
// pad_interval/pad_amount codes: pad (pad_amount+1) DWORDs after every
// 2^(pad_interval+1) DWORDs stored -> reproduces the padded LDS strides.
__device__ __forceinline__ void tdm_load_2d_f16(
    void* lds_dst, const void* gsrc,
    unsigned tile_d0, unsigned tile_d1,
    unsigned long long row_stride_elems,
    unsigned pad_interval_code, unsigned pad_amount_code)
{
    const unsigned long long ga = (unsigned long long)gsrc;
    u32x4 g0;
    g0[0] = 1u;                                       // count=1, user mode
    g0[1] = (unsigned)(unsigned long long)lds_dst;    // lds_addr
    g0[2] = (unsigned)ga;                             // global_addr[31:0]
    g0[3] = ((unsigned)(ga >> 32) & 0x01FFFFFFu) | (2u << 30);  // addr hi, type=2

    i32x8 g1;
    g1[0] = (int)((1u << 16) |                        // data_size = 2 bytes
                  (1u << 20) |                        // pad_enable
                  (pad_interval_code << 22) |
                  (pad_amount_code << 25));
    g1[1] = (int)0xFFFF0000u;                         // tensor_dim0 lo16 (large)
    g1[2] = (int)0xFFFF7FFFu;                         // dim0 hi16 / dim1 lo16
    g1[3] = (int)((tile_d0 << 16) | 0x7FFFu);         // dim1 hi16 / tile_dim0
    g1[4] = (int)(tile_d1 & 0xFFFFu);                 // tile_dim1 (tile_dim2=0)
    g1[5] = (int)(unsigned)(row_stride_elems & 0xFFFFFFFFull);  // dim0_stride lo
    g1[6] = (int)(unsigned)((row_stride_elems >> 32) & 0xFFFFull);
    g1[7] = 0;

    const i32x4 z4 = {0, 0, 0, 0};
#if __has_include(<hip/amd_detail/amd_gfx1250_TDM.h>)
    const i32x8 z8 = {0, 0, 0, 0, 0, 0, 0, 0};
    __builtin_amdgcn_tensor_load_to_lds(g0, g1, z4, z4, z8, 0);
#else
    __builtin_amdgcn_tensor_load_to_lds(g0, g1, z4, z4, 0);
#endif
}

// A-operand fragment (16x32 f16). Lane L (m=L&15, half=L>>4) holds
// k = half*16 + j: two contiguous
// 16-byte chunks per lane from a row-major tile with stride lda (elements).
__device__ __forceinline__ v16h load_a_rm(const f16* p, int lda) {
    const int lane = threadIdx.x & 31;
    const int m = lane & 15, hl = lane >> 4;
    const f16* r = p + m * lda + hl * 16;
    return pack2(*(const v8h*)(r), *(const v8h*)(r + 8));
}

// B-operand fragment (32x16 f16). Lane L (n=L&15, kh=L>>4) holds
// k = kh*16 + j -> one strided column read from a row-major KxN tile.
__device__ __forceinline__ v16h load_b_rm(const f16* p, int ldb) {
    const int lane = threadIdx.x & 31;
    const int n = lane & 15, kh = lane >> 4;
    const f16* c = p + kh * 16 * ldb + n;
    v16h o;
#pragma unroll
    for (int j = 0; j < 16; ++j) o[j] = c[j * ldb];
    return o;
}

// ---------------------------------------------------------------------------
// f32 -> f16 weight conversion
__global__ void cvt_f32_f16(const float* __restrict__ in, f16* __restrict__ out, int n) {
    for (int i = blockIdx.x * blockDim.x + threadIdx.x; i < n; i += gridDim.x * blockDim.x)
        out[i] = (f16)in[i];
}

// ---------------------------------------------------------------------------
// LayerNorm (one block per row) producing f16 activations
__global__ __launch_bounds__(256) void ln_f16_kernel(
    const float* __restrict__ x, const float* __restrict__ g,
    const float* __restrict__ be, f16* __restrict__ out, int D)
{
    __shared__ float s1[256], s2[256];
    const int row = blockIdx.x;
    const float* xr = x + (size_t)row * D;
    float s = 0.0f, ss = 0.0f;
    for (int i = threadIdx.x; i < D; i += 256) { float v = xr[i]; s += v; ss += v * v; }
    s1[threadIdx.x] = s; s2[threadIdx.x] = ss;
    __syncthreads();
    for (int st = 128; st > 0; st >>= 1) {
        if (threadIdx.x < st) {
            s1[threadIdx.x] += s1[threadIdx.x + st];
            s2[threadIdx.x] += s2[threadIdx.x + st];
        }
        __syncthreads();
    }
    const float mu = s1[0] / D;
    const float var = s2[0] / D - mu * mu;
    const float rs = rsqrtf(var + 1e-5f);
    f16* orow = out + (size_t)row * D;
    for (int i = threadIdx.x; i < D; i += 256)
        orow[i] = (f16)((xr[i] - mu) * rs * g[i] + be[i]);
}

// ---------------------------------------------------------------------------
// WMMA GEMM: out = epilogue(A[MxK] @ Bw[KxN] + bias)
// MODE 0: out16 = fpq(acc+bias)
// MODE 1: out16 = relu(fpq(acc+bias))
// MODE 2: out32 = fpq(res + fpq(acc+bias))
// Block tile 128x128, 8 waves (2M x 4N), each wave 4x2 16x16 tiles, K-tile 32.
// Tiles are double-buffered in LDS and filled by the Tensor Data Mover:
// wave 0 issues two descriptor-driven 2D tile DMAs per K-step (TENSORcnt),
// overlapping the next tile's traffic with the current WMMA work.
template <int MODE>
__global__ __launch_bounds__(256) void gemm_wmma(
    const f16* __restrict__ A, const f16* __restrict__ Bw,
    const float* __restrict__ bias, const float* __restrict__ res,
    f16* __restrict__ out16, float* __restrict__ out32,
    int M, int N, int K)
{
    __shared__ __align__(16) f16 As[2][128][40];   // 32-elem rows + 8 pad (TDM pad)
    __shared__ __align__(16) f16 Bs[2][32][136];   // 128-elem rows + 8 pad (TDM pad)

    const int tid = threadIdx.x;
    const int wave = tid >> 5;
    const int wm = wave & 1;   // 2 waves across M
    const int wn = wave >> 1;  // 4 waves across N
    const int m0 = blockIdx.y * 128;
    const int n0 = blockIdx.x * 128;

    const v8f vzero = {};
    v8f acc[4][2];
#pragma unroll
    for (int i = 0; i < 4; ++i)
#pragma unroll
        for (int j = 0; j < 2; ++j) acc[i][j] = vzero;

    auto stage = [&](int buf, int kt) {
        // A tile: 128 rows x 32 f16, row stride K; pad 4 DW every 16 DW -> ld 40
        tdm_load_2d_f16(&As[buf][0][0], A + (size_t)m0 * K + kt,
                        32u, 128u, (unsigned long long)K, 3u, 3u);
        // B tile: 32 rows x 128 f16, row stride N; pad 4 DW every 64 DW -> ld 136
        tdm_load_2d_f16(&Bs[buf][0][0], Bw + (size_t)kt * N + n0,
                        128u, 32u, (unsigned long long)N, 5u, 3u);
    };

    if (wave == 0) {
        stage(0, 0);
        __builtin_amdgcn_s_wait_tensorcnt(0);
    }
    __syncthreads();

    int buf = 0;
    for (int kt = 0; kt < K; kt += 32) {
        if (wave == 0 && kt + 32 < K) stage(buf ^ 1, kt + 32);  // prefetch

        v16h bf[2];
#pragma unroll
        for (int nt = 0; nt < 2; ++nt)
            bf[nt] = load_b_rm(&Bs[buf][0][wn * 32 + nt * 16], 136);
#pragma unroll
        for (int mt = 0; mt < 4; ++mt) {
            v16h af = load_a_rm(&As[buf][wm * 64 + mt * 16][0], 40);
#pragma unroll
            for (int nt = 0; nt < 2; ++nt)
                acc[mt][nt] = wmma16(af, bf[nt], acc[mt][nt]);
        }
        if (wave == 0) __builtin_amdgcn_s_wait_tensorcnt(0);  // DMAs landed
        __syncthreads();   // release: buf^1 ready, reads of buf done
        buf ^= 1;
    }

    const int lane = tid & 31;
    const int hl = lane >> 4, nn = lane & 15;
#pragma unroll
    for (int mt = 0; mt < 4; ++mt) {
#pragma unroll
        for (int nt = 0; nt < 2; ++nt) {
#pragma unroll
            for (int r = 0; r < 8; ++r) {
                const int row = m0 + wm * 64 + mt * 16 + hl * 8 + r;
                const int col = n0 + wn * 32 + nt * 16 + nn;
                float v = fpq1(acc[mt][nt][r] + bias[col]);
                if (MODE == 1) v = fmaxf(v, 0.0f);
                if (MODE == 2) {
                    out32[(size_t)row * N + col] = fpq1(res[(size_t)row * N + col] + v);
                } else {
                    out16[(size_t)row * N + col] = (f16)v;
                }
            }
        }
    }
}

// ---------------------------------------------------------------------------
// Attention: one block per (q-block of 64 rows, head, batch); 4 waves, each
// wave owns 16 query rows. Two passes: (1) online softmax stats via streamed
// QK^T WMMA tiles, (2) recompute scores, fpq(softmax), PV WMMA.
__global__ __launch_bounds__(128) void attn_kernel(
    const f16* __restrict__ qkv, f16* __restrict__ ctx, int S, int H, int hd)
{
    const int Dq = 3 * H * hd;  // 3072
    const int D = H * hd;       // 1024

    __shared__ __align__(16) f16 Vs[32][72];
    __shared__ __align__(16) f16 Ps[4][16][40];

    const int tid = threadIdx.x;
    const int w = tid >> 5;
    const int lane = tid & 31;
    const int hl = lane >> 4, nn = lane & 15;

    const int qb = blockIdx.x, h = blockIdx.y, b = blockIdx.z;
    const int qrow0 = qb * 64 + w * 16;

    // Q fragments in A layout directly from global (row = query, k = d)
    v16h qa[2];
    {
        const f16* base = qkv + (size_t)(b * S + qrow0 + nn) * Dq + h * hd;
#pragma unroll
        for (int kc = 0; kc < 2; ++kc)
            qa[kc] = pack2(*(const v8h*)(base + kc * 32 + hl * 16),
                           *(const v8h*)(base + kc * 32 + hl * 16 + 8));
    }

    const float scale = rsqrtf((float)hd);
    const v8f vzero = {};

    float rm[8], rl[8];
#pragma unroll
    for (int r = 0; r < 8; ++r) { rm[r] = -1e30f; rl[r] = 0.0f; }

    // ---- pass 1: per-lane online max / sum-exp over all key tiles ----
    for (int st = 0; st < S / 16; ++st) {
        const f16* kb = qkv + (size_t)(b * S + st * 16 + nn) * Dq + D + h * hd + hl * 16;
        v16h b0 = pack2(*(const v8h*)(kb),      *(const v8h*)(kb + 8));
        v16h b1 = pack2(*(const v8h*)(kb + 32), *(const v8h*)(kb + 40));
        v8f sa = vzero;
        sa = wmma16(qa[0], b0, sa);
        sa = wmma16(qa[1], b1, sa);
#pragma unroll
        for (int r = 0; r < 8; ++r) {
            const float sv = sa[r] * scale;
            const float nm = fmaxf(rm[r], sv);
            rl[r] = rl[r] * __expf(rm[r] - nm) + __expf(sv - nm);
            rm[r] = nm;
        }
    }
    // combine across the 16 lanes holding one row (xor stays within half-wave)
#pragma unroll
    for (int r = 0; r < 8; ++r) {
#pragma unroll
        for (int off = 1; off < 16; off <<= 1) {
            const float om = __shfl_xor(rm[r], off, 32);
            const float ol = __shfl_xor(rl[r], off, 32);
            const float nm = fmaxf(rm[r], om);
            rl[r] = rl[r] * __expf(rm[r] - nm) + ol * __expf(om - nm);
            rm[r] = nm;
        }
        rl[r] = 1.0f / rl[r];  // reciprocal of denominator
    }

    // ---- pass 2: probs = fpq(softmax), ctx += P @ V ----
    v8f oacc[4];
#pragma unroll
    for (int nt = 0; nt < 4; ++nt) oacc[nt] = vzero;

    const int vrow = tid >> 2, vseg = tid & 3;

    for (int ks = 0; ks < S / 32; ++ks) {
        __syncthreads();   // everyone done reading previous V tile
        {   // async stage of V[ks*32 .. +32][hd] into LDS (ASYNCcnt path)
            const f16* vg = qkv + (size_t)(b * S + ks * 32 + vrow) * Dq + 2 * D +
                            h * hd + vseg * 16;
            async_ld16(&Vs[vrow][vseg * 16], vg);
            async_ld16(&Vs[vrow][vseg * 16 + 8], vg + 8);
        }
        wait_async0();
        __syncthreads();

        // recompute the two 16x16 score subtiles and park probs in LDS
#pragma unroll
        for (int sub = 0; sub < 2; ++sub) {
            const int st = ks * 2 + sub;
            const f16* kb = qkv + (size_t)(b * S + st * 16 + nn) * Dq + D +
                            h * hd + hl * 16;
            v16h b0 = pack2(*(const v8h*)(kb),      *(const v8h*)(kb + 8));
            v16h b1 = pack2(*(const v8h*)(kb + 32), *(const v8h*)(kb + 40));
            v8f sa = vzero;
            sa = wmma16(qa[0], b0, sa);
            sa = wmma16(qa[1], b1, sa);
#pragma unroll
            for (int r = 0; r < 8; ++r) {
                const float p = fpq1(__expf(sa[r] * scale - rm[r]) * rl[r]);
                Ps[w][hl * 8 + r][sub * 16 + nn] = (f16)p;
            }
        }
        // wave-private C-layout -> A-layout bounce; DS ops are in-order per
        // wave but force completion before the cross-lane re-read
        asm volatile("s_wait_dscnt 0" ::: "memory");

        v16h pa = load_a_rm(&Ps[w][0][0], 40);
#pragma unroll
        for (int nt = 0; nt < 4; ++nt) {
            v16h vb = load_b_rm(&Vs[0][nt * 16], 72);
            oacc[nt] = wmma16(pa, vb, oacc[nt]);
        }
    }

#pragma unroll
    for (int nt = 0; nt < 4; ++nt) {
#pragma unroll
        for (int r = 0; r < 8; ++r) {
            const int row = b * S + qrow0 + hl * 8 + r;
            ctx[(size_t)row * D + h * hd + nt * 16 + nn] = (f16)oacc[nt][r];
        }
    }
}

// ---------------------------------------------------------------------------
extern "C" void kernel_launch(void* const* d_in, const int* in_sizes, int n_in,
                              void* d_out, int out_size, void* d_ws, size_t ws_size,
                              hipStream_t stream)
{
    const float* x    = (const float*)d_in[0];
    const float* Wqkv = (const float*)d_in[1];
    const float* bqkv = (const float*)d_in[2];
    const float* Wout = (const float*)d_in[3];
    const float* bout = (const float*)d_in[4];
    const float* W1   = (const float*)d_in[5];
    const float* b1   = (const float*)d_in[6];
    const float* W2   = (const float*)d_in[7];
    const float* b2   = (const float*)d_in[8];
    const float* g1   = (const float*)d_in[9];
    const float* be1  = (const float*)d_in[10];
    const float* g2   = (const float*)d_in[11];
    const float* be2  = (const float*)d_in[12];
    (void)in_sizes; (void)n_in; (void)out_size; (void)ws_size;

    const int Bq = 8, S = 1024, D = 1024, H = 4096, NH = 16, hd = 64;
    const int M = Bq * S;

    char* p = (char*)d_ws;
    auto alloc = [&](size_t bytes) {
        char* r = p; p += (bytes + 255) & ~(size_t)255; return r;
    };
    f16*   w16_qkv = (f16*)alloc((size_t)D * 3 * D * 2);
    f16*   w16_out = (f16*)alloc((size_t)D * D * 2);
    f16*   w16_1   = (f16*)alloc((size_t)D * H * 2);
    f16*   w16_2   = (f16*)alloc((size_t)H * D * 2);
    f16*   xn16    = (f16*)alloc((size_t)M * 3 * D * 2);  // reused: xn then xn2
    f16*   qkv16   = (f16*)alloc((size_t)M * 3 * D * 2);
    f16*   ctx16   = (f16*)alloc((size_t)M * D * 2);
    float* x2      = (float*)alloc((size_t)M * D * 4);
    f16*   h16     = (f16*)alloc((size_t)M * H * 2);

    cvt_f32_f16<<<2048, 256, 0, stream>>>(Wqkv, w16_qkv, D * 3 * D);
    cvt_f32_f16<<<2048, 256, 0, stream>>>(Wout, w16_out, D * D);
    cvt_f32_f16<<<2048, 256, 0, stream>>>(W1,   w16_1,   D * H);
    cvt_f32_f16<<<2048, 256, 0, stream>>>(W2,   w16_2,   H * D);

    // xn = LN(x)
    ln_f16_kernel<<<M, 256, 0, stream>>>(x, g1, be1, xn16, D);

    // qkv = fpq(xn @ Wqkv + bqkv)
    gemm_wmma<0><<<dim3(3 * D / 128, M / 128), 256, 0, stream>>>(
        xn16, w16_qkv, bqkv, nullptr, qkv16, nullptr, M, 3 * D, D);

    // ctx = softmax-attention(q,k,v) with fpq'd probabilities
    attn_kernel<<<dim3(S / 64, NH, Bq), 128, 0, stream>>>(qkv16, ctx16, S, NH, hd);

    // x2 = fpq(x + fpq(ctx @ Wout + bout))
    gemm_wmma<2><<<dim3(D / 128, M / 128), 256, 0, stream>>>(
        ctx16, w16_out, bout, x, nullptr, x2, M, D, D);

    // xn2 = LN(x2)
    f16* xn2_16 = xn16;
    ln_f16_kernel<<<M, 256, 0, stream>>>(x2, g2, be2, xn2_16, D);

    // h = relu(fpq(xn2 @ W1 + b1))
    gemm_wmma<1><<<dim3(H / 128, M / 128), 256, 0, stream>>>(
        xn2_16, w16_1, b1, nullptr, h16, nullptr, M, H, D);

    // out = fpq(x2 + fpq(h @ W2 + b2))
    gemm_wmma<2><<<dim3(D / 128, M / 128), 256, 0, stream>>>(
        h16, w16_2, b2, x2, nullptr, (float*)d_out, M, D, H);
}
